// MultiHeadAttention_layer_7464653160482
// MI455X (gfx1250) — compile-verified, matrix operand order checked by reading, not run on hardware
//
#include <hip/hip_runtime.h>
#include <hip/hip_bf16.h>
#include <math.h>

// MI455X / gfx1250, wave32. bf16 WMMA 16x16x32, fp32 accumulate.
typedef __attribute__((ext_vector_type(16))) __bf16 v16bf;
typedef __attribute__((ext_vector_type(8)))  __bf16 v8bf;
typedef __attribute__((ext_vector_type(8)))  float  v8f;
typedef __attribute__((ext_vector_type(4)))  float  v4f;
typedef __attribute__((ext_vector_type(4)))  unsigned int v4u;
typedef __attribute__((ext_vector_type(8)))  int    v8i;
typedef __attribute__((ext_vector_type(4)))  int    v4i;

#define B_      4
#define S_      2048
#define D_      1024
#define H_      16
#define DEPTH_  64

union bf16x16 { v16bf v; v8bf h[2]; };

// 16-bit WMMA A/B operand (16x32): lane half 0 holds K pairs {0..7, 16..23},
// half 1 holds {8..15, 24..31}  ->  two contiguous 16B runs per lane.
__device__ __forceinline__ v16bf frag_bf16(const __bf16* row, int kk, int half) {
    bf16x16 u;
    u.h[0] = *(const v8bf*)(row + kk + 8 * half);
    u.h[1] = *(const v8bf*)(row + kk + 16 + 8 * half);
    return u.v;
}

// Same operand built from fp32 data (global or LDS): 4x b128 loads + cvt_pk.
__device__ __forceinline__ v16bf frag_f32(const float* row, int kk, int half) {
    const float* p = row + kk + 8 * half;
    v4f a0 = *(const v4f*)(p);
    v4f a1 = *(const v4f*)(p + 4);
    v4f a2 = *(const v4f*)(p + 16);
    v4f a3 = *(const v4f*)(p + 20);
    bf16x16 u;
#pragma unroll
    for (int i = 0; i < 4; ++i) {
        u.v[i]      = (__bf16)a0[i];
        u.v[4 + i]  = (__bf16)a1[i];
        u.v[8 + i]  = (__bf16)a2[i];
        u.v[12 + i] = (__bf16)a3[i];
    }
    return u.v;
}

#define WMMA_BF16(a, b, c) \
    __builtin_amdgcn_wmma_f32_16x16x32_bf16(false, (a), false, (b), (short)0, (c), false, false)

// ---------------------------------------------------------------------------
// TDM: async 2D bf16 tile load Global -> LDS (Tensor Data Mover, TENSORcnt).
// Descriptor layout per CDNA5 ISA 8.3/8.4. Arity differs per toolchain.
// ---------------------------------------------------------------------------
__device__ __forceinline__ void tdm_load_2d_bf16(unsigned lds_off,
                                                 const void* gaddr,
                                                 unsigned tile_rows,   // tile_dim1
                                                 unsigned row_elems,   // tile_dim0
                                                 unsigned row_stride,  // tensor_dim0_stride
                                                 unsigned tensor_rows) // tensor_dim1 (OOB)
{
    unsigned long long ga = (unsigned long long)(size_t)gaddr;
    v4u g0 = {};
    g0[0] = 1u;                                          // count=1 (valid), user mode
    g0[1] = lds_off;                                     // lds_addr (bytes)
    g0[2] = (unsigned)(ga & 0xffffffffu);                // global_addr[31:0]
    g0[3] = (unsigned)((ga >> 32) & 0x1ffffffu) | (2u << 30); // addr[56:32], type=2
    v8i g1 = {};
    g1[0] = (int)(1u << 16);                             // wg_mask=0, data_size=1 (2B)
    g1[1] = (int)((row_elems & 0xffffu) << 16);          // tensor_dim0[15:0]
    g1[2] = (int)(((row_elems >> 16) & 0xffffu) |        // tensor_dim0[31:16]
                  ((tensor_rows & 0xffffu) << 16));      // tensor_dim1[15:0]
    g1[3] = (int)(((tensor_rows >> 16) & 0xffffu) |      // tensor_dim1[31:16]
                  ((row_elems & 0xffffu) << 16));        // tile_dim0
    g1[4] = (int)(tile_rows & 0xffffu);                  // tile_dim1 (tile_dim2=0)
    g1[5] = (int)row_stride;                             // tensor_dim0_stride[31:0]
    g1[6] = 0;
    g1[7] = 0;
    v4i z4 = {};
#if defined(__clang_major__) && (__clang_major__ >= 23)
    v8i z8 = {};
    __builtin_amdgcn_tensor_load_to_lds(g0, g1, z4, z4, z8, 0);
#else
    __builtin_amdgcn_tensor_load_to_lds(g0, g1, z4, z4, 0);
#endif
}

// ---------------------------------------------------------------------------
// Stage 0: W (fp32, [k][n]) -> Wt (bf16, [n][k]) for all four weights.
// ---------------------------------------------------------------------------
__global__ void mha_wt_bf16(const float* __restrict__ Wq, const float* __restrict__ Wk,
                            const float* __restrict__ Wv, const float* __restrict__ Wo,
                            __bf16* __restrict__ WqT, __bf16* __restrict__ WkT,
                            __bf16* __restrict__ WvT, __bf16* __restrict__ WoT)
{
    __shared__ float tile[32][33];
    const int which = blockIdx.z;
    const float* W = (which == 0) ? Wq : (which == 1) ? Wk : (which == 2) ? Wv : Wo;
    __bf16*      T = (which == 0) ? WqT : (which == 1) ? WkT : (which == 2) ? WvT : WoT;
    const int k0 = blockIdx.y * 32;
    const int n0 = blockIdx.x * 32;
    const int tx = threadIdx.x;   // 32
    const int ty = threadIdx.y;   // 8
    for (int i = ty; i < 32; i += 8)
        tile[i][tx] = W[(size_t)(k0 + i) * D_ + n0 + tx];
    __syncthreads();
    for (int i = ty; i < 32; i += 8)
        T[(size_t)(n0 + i) * D_ + k0 + tx] = (__bf16)tile[tx][i];
}

// ---------------------------------------------------------------------------
// Stage 1: Q/K/V projections. Each wave computes a 16x64 strip (A reused x4).
// Q,K written split-head [B,H,S,64] bf16; V written transposed [B,H,64,S].
// ---------------------------------------------------------------------------
__global__ void mha_proj_qkv(const float* __restrict__ q,
                             const float* __restrict__ k,
                             const float* __restrict__ v,
                             const __bf16* __restrict__ WqT, const float* __restrict__ bq,
                             const __bf16* __restrict__ WkT, const float* __restrict__ bk,
                             const __bf16* __restrict__ WvT, const float* __restrict__ bv,
                             __bf16* __restrict__ Qo, __bf16* __restrict__ Ko,
                             __bf16* __restrict__ Vt)
{
    const int which = blockIdx.y;
    const float*  X    = (which == 0) ? q   : (which == 1) ? k   : v;
    const __bf16* Wt   = (which == 0) ? WqT : (which == 1) ? WkT : WvT;
    const float*  bias = (which == 0) ? bq  : (which == 1) ? bk  : bv;
    __bf16* Out        = (which == 0) ? Qo  : Ko;   // unused when which==2

    const int lane = threadIdx.x & 31;
    const int wave = threadIdx.x >> 5;
    const int job  = blockIdx.x * 8 + wave;         // 512 Mtiles * 16 Nstrips
    const int m0   = (job / 16) * 16;
    const int n0s  = (job % 16) * 64;
    const int half = lane >> 4;
    const int ml   = lane & 15;

    const float* arow = X + (size_t)(m0 + ml) * D_;
    v8f acc[4] = {};
    for (int kk = 0; kk < D_; kk += 32) {
        __builtin_prefetch(arow + kk + 128, 0, 0);          // global_prefetch_b8
        const v16bf a = frag_f32(arow, kk, half);
#pragma unroll
        for (int t = 0; t < 4; ++t) {
            const v16bf b = frag_bf16(Wt + (size_t)(n0s + t * 16 + ml) * D_, kk, half);
            acc[t] = WMMA_BF16(a, b, acc[t]);
        }
    }

#pragma unroll
    for (int t = 0; t < 4; ++t)
#pragma unroll
        for (int r = 0; r < 8; ++r) {
            const int row = m0 + r + 8 * half;              // [0, B*S)
            const int col = n0s + t * 16 + ml;              // [0, D)
            const float val = acc[t][r] + bias[col];
            const int bi = row / S_, s = row % S_;
            const int h  = col / DEPTH_, d = col % DEPTH_;
            if (which == 2)
                Vt[((size_t)(bi * H_ + h) * DEPTH_ + d) * S_ + s] = (__bf16)val;
            else
                Out[(((size_t)bi * H_ + h) * S_ + s) * DEPTH_ + d] = (__bf16)val;
        }
}

// ---------------------------------------------------------------------------
// Stage 2: fused attention per (b*h, 16-query tile). 8 waves.
// K panel streamed by the Tensor Data Mover, DOUBLE-BUFFERED: wave0 issues
// chunk c+1, waits tensorcnt<=1 (TDM completes in-order), workgroup computes
// chunk c. logits -> LDS -> wave-shuffle softmax (float4) -> one b128 attn
// write -> attn@V.
// LDS: 16*2048 f32 (128K) + 8*4*16*16 f32 (32K) + 2x 256*64 bf16 (64K) = 224K.
// ---------------------------------------------------------------------------
#define KSTAGE_OFF (16 * S_ * 4 + 8 * 4 * 16 * 16 * 4)      // 163840
#define KCHUNK     256
#define KBYTES     (KCHUNK * DEPTH_ * 2)                    // 32768
#define NCHUNK     (S_ / KCHUNK)                            // 8

__global__ void mha_attn(const __bf16* __restrict__ Q,
                         const __bf16* __restrict__ K,
                         const __bf16* __restrict__ Vt,     // [B*H, 64, S]
                         const float* __restrict__ mask,
                         float* __restrict__ attn_out,      // [B*H, S, S]
                         __bf16* __restrict__ ctx)          // [B*S, D]
{
    extern __shared__ char smem_raw[];
    float*  attn_s = (float*)smem_raw;                         // 16 x S_
    float*  part_s = (float*)(smem_raw + 16 * S_ * 4);         // 8 x 4 x 16 x 16

    const int bh = blockIdx.x;
    const int q0 = blockIdx.y * 16;
    const int b  = bh / H_;
    const int h  = bh % H_;

    const int lane = threadIdx.x & 31;
    const int wave = threadIdx.x >> 5;
    const int half = lane >> 4;
    const int ml   = lane & 15;

    const __bf16* Qb  = Q  + (size_t)bh * S_ * DEPTH_;
    const __bf16* Kb  = K  + (size_t)bh * S_ * DEPTH_;
    const __bf16* Vtb = Vt + (size_t)bh * DEPTH_ * S_;

    // Q tile 16x64 = two A operands, reused across all key tiles.
    const __bf16* qrow = Qb + (size_t)(q0 + ml) * DEPTH_;
    const v16bf aq0 = frag_bf16(qrow, 0, half);
    const v16bf aq1 = frag_bf16(qrow, 32, half);

    // ---- logits = (Q K^T)/8 + mask*-1e9, K streamed by double-buffered TDM
    const float scale = 0.125f;
    if (wave == 0)
        tdm_load_2d_bf16(KSTAGE_OFF, Kb, KCHUNK, DEPTH_, DEPTH_, S_);
    for (int c = 0; c < NCHUNK; ++c) {
        const int cur = c & 1;
        if (wave == 0) {
            if (c + 1 < NCHUNK) {
                tdm_load_2d_bf16(KSTAGE_OFF + (1 - cur) * KBYTES,
                                 Kb + (size_t)(c + 1) * KCHUNK * DEPTH_,
                                 KCHUNK, DEPTH_, DEPTH_, S_ - (c + 1) * KCHUNK);
                __builtin_amdgcn_s_wait_tensorcnt(1);   // chunk c done (in-order)
            } else {
                __builtin_amdgcn_s_wait_tensorcnt(0);
            }
        }
        __syncthreads();                                 // chunk c visible
        const __bf16* ks = (const __bf16*)(smem_raw + KSTAGE_OFF + cur * KBYTES);
#pragma unroll
        for (int t = 0; t < 2; ++t) {
            const int ktile = wave * 2 + t;              // [0,16) within chunk
            const __bf16* krow = ks + (ktile * 16 + ml) * DEPTH_;
            const v16bf b0 = frag_bf16(krow, 0, half);
            const v16bf b1 = frag_bf16(krow, 32, half);
            v8f cacc = {};
            cacc = WMMA_BF16(aq0, b0, cacc);
            cacc = WMMA_BF16(aq1, b1, cacc);
            const int col0 = c * KCHUNK + ktile * 16;
#pragma unroll
            for (int r = 0; r < 8; ++r) {
                const int row = r + 8 * half;
                const float mval = mask[(size_t)(q0 + row) * S_ + col0 + ml];
                attn_s[row * S_ + col0 + ml] = cacc[r] * scale + mval * -1e9f;
            }
        }
        __syncthreads();                                 // buffer reusable
    }

    // ---- softmax: wave w owns rows 2w, 2w+1; float4 + wave32 shuffles -----
#pragma unroll
    for (int rr = 0; rr < 2; ++rr) {
        const int row = wave * 2 + rr;
        v4f* arow4 = (v4f*)(attn_s + row * S_);
        float mx = -3.0e38f;
        for (int i = lane; i < S_ / 4; i += 32) {
            const v4f x = arow4[i];
            mx = fmaxf(mx, fmaxf(fmaxf(x[0], x[1]), fmaxf(x[2], x[3])));
        }
#pragma unroll
        for (int off = 16; off > 0; off >>= 1) mx = fmaxf(mx, __shfl_xor(mx, off, 32));
        float sum = 0.f;
        for (int i = lane; i < S_ / 4; i += 32) {
            const v4f x = arow4[i];
            v4f e;
#pragma unroll
            for (int j = 0; j < 4; ++j) e[j] = __expf(x[j] - mx);
            arow4[i] = e;
            sum += e[0] + e[1] + e[2] + e[3];
        }
#pragma unroll
        for (int off = 16; off > 0; off >>= 1) sum += __shfl_xor(sum, off, 32);
        const float inv = 1.0f / sum;
        v4f* gout4 = (v4f*)(attn_out + ((size_t)bh * S_ + (q0 + row)) * S_);
        for (int i = lane; i < S_ / 4; i += 32) {
            const v4f p = arow4[i] * inv;
            arow4[i] = p;          // keep probs in LDS for attn@V
            gout4[i] = p;          // the one HBM write of attn (roofline term)
        }
    }
    __syncthreads();

    // ---- ctx = attn @ V : M=16, N=64, K=2048; wave w covers 256 keys ------
    v8f acc[4] = {};
    const int kbase = wave * 256;
    for (int kk = kbase; kk < kbase + 256; kk += 32) {
        const v16bf a = frag_f32(attn_s + ml * S_, kk, half);   // LDS b128 + cvt
#pragma unroll
        for (int dt = 0; dt < 4; ++dt) {
            const v16bf bv = frag_bf16(Vtb + (size_t)(dt * 16 + ml) * S_, kk, half);
            acc[dt] = WMMA_BF16(a, bv, acc[dt]);
        }
    }
#pragma unroll
    for (int dt = 0; dt < 4; ++dt)
#pragma unroll
        for (int r = 0; r < 8; ++r)
            part_s[(((wave * 4 + dt) * 16) + (r + 8 * half)) * 16 + ml] = acc[dt][r];
    __syncthreads();

    for (int o = threadIdx.x; o < 16 * DEPTH_; o += 256) {
        const int m = o / DEPTH_;
        const int d = o % DEPTH_;
        const int dt = d / 16, n = d % 16;
        float s = 0.f;
#pragma unroll
        for (int w = 0; w < 8; ++w) s += part_s[(((w * 4 + dt) * 16) + m) * 16 + n];
        ctx[((size_t)b * S_ + (q0 + m)) * D_ + h * DEPTH_ + d] = (__bf16)s;
    }
}

// ---------------------------------------------------------------------------
// Stage 3: out = ctx @ Wo + bo. 16x64 strip per wave, all-b128 operands.
// ---------------------------------------------------------------------------
__global__ void mha_out_proj(const __bf16* __restrict__ ctx,
                             const __bf16* __restrict__ WoT,
                             const float* __restrict__ bo,
                             float* __restrict__ out)
{
    const int lane = threadIdx.x & 31;
    const int wave = threadIdx.x >> 5;
    const int job  = blockIdx.x * 8 + wave;
    const int m0   = (job / 16) * 16;
    const int n0s  = (job % 16) * 64;
    const int half = lane >> 4;
    const int ml   = lane & 15;

    const __bf16* arow = ctx + (size_t)(m0 + ml) * D_;
    v8f acc[4] = {};
    for (int kk = 0; kk < D_; kk += 32) {
        __builtin_prefetch(arow + kk + 256, 0, 0);
        const v16bf a = frag_bf16(arow, kk, half);
#pragma unroll
        for (int t = 0; t < 4; ++t) {
            const v16bf b = frag_bf16(WoT + (size_t)(n0s + t * 16 + ml) * D_, kk, half);
            acc[t] = WMMA_BF16(a, b, acc[t]);
        }
    }
#pragma unroll
    for (int t = 0; t < 4; ++t)
#pragma unroll
        for (int r = 0; r < 8; ++r) {
            const int row = m0 + r + 8 * half;
            const int col = n0s + t * 16 + ml;
            out[(size_t)row * D_ + col] = acc[t][r] + bo[col];
        }
}

// ---------------------------------------------------------------------------
extern "C" void kernel_launch(void* const* d_in, const int* in_sizes, int n_in,
                              void* d_out, int out_size, void* d_ws, size_t ws_size,
                              hipStream_t stream)
{
    const float* q    = (const float*)d_in[0];
    const float* k    = (const float*)d_in[1];
    const float* v    = (const float*)d_in[2];
    const float* mask = (const float*)d_in[3];
    const float* Wq   = (const float*)d_in[4];
    const float* bq   = (const float*)d_in[5];
    const float* Wk   = (const float*)d_in[6];
    const float* bk   = (const float*)d_in[7];
    const float* Wv   = (const float*)d_in[8];
    const float* bv   = (const float*)d_in[9];
    const float* Wo   = (const float*)d_in[10];
    const float* bo   = (const float*)d_in[11];

    float* out  = (float*)d_out;                               // [B,S,D]
    float* attn = out + (size_t)B_ * S_ * D_;                  // [B,H,S,S]

    const size_t elems = (size_t)B_ * S_ * D_;                 // 8M
    __bf16* Qo  = (__bf16*)d_ws;                               // [B,H,S,64]
    __bf16* Ko  = Qo + elems;                                  // [B,H,S,64]
    __bf16* Vt  = Ko + elems;                                  // [B,H,64,S]
    __bf16* ctx = Vt + elems;                                  // [B*S, D]
    __bf16* WqT = ctx + elems;                                 // 4x [D,D] bf16
    __bf16* WkT = WqT + (size_t)D_ * D_;
    __bf16* WvT = WkT + (size_t)D_ * D_;
    __bf16* WoT = WvT + (size_t)D_ * D_;
    // ws use: 4*16MB + 4*2MB = 72 MB

    // Stage 0: transpose+convert weights once
    {
        dim3 grid(D_ / 32, D_ / 32, 4);
        mha_wt_bf16<<<grid, dim3(32, 8), 0, stream>>>(Wq, Wk, Wv, Wo,
                                                      WqT, WkT, WvT, WoT);
    }
    // Stage 1: projections (512 Mtiles * 16 Nstrips / 8 waves, y = q|k|v)
    {
        dim3 grid((B_ * S_ / 16) * (D_ / 64) / 8, 3);          // (1024, 3)
        mha_proj_qkv<<<grid, 256, 0, stream>>>(q, k, v, WqT, bq, WkT, bk, WvT, bv,
                                               Qo, Ko, Vt);
    }
    // Stage 2: fused attention (double-buffered TDM-staged K)
    {
        const int smem = KSTAGE_OFF + 2 * KBYTES;              // 229376
        (void)hipFuncSetAttribute(reinterpret_cast<const void*>(mha_attn),
                                  hipFuncAttributeMaxDynamicSharedMemorySize, smem);
        dim3 grid(B_ * H_, S_ / 16);                           // (64, 128)
        mha_attn<<<grid, 256, smem, stream>>>(Qo, Ko, Vt, mask, attn, ctx);
    }
    // Stage 3: output projection
    {
        dim3 grid((B_ * S_ / 16) * (D_ / 64) / 8);             // 1024
        mha_out_proj<<<grid, 256, 0, stream>>>(ctx, WoT, bo, out);
    }
}
